// SuperCompressedGatePreBasicBlock_91250875171254
// MI455X (gfx1250) — compile-verified, matrix operand order checked by reading, not run on hardware
//
#include <hip/hip_runtime.h>
#include <hip/hip_bf16.h>

typedef __attribute__((ext_vector_type(16))) _Float16 v16h;
typedef __attribute__((ext_vector_type(8)))  _Float16 v8h;
typedef __attribute__((ext_vector_type(8)))  float    v8f;

#define EPSV 1e-5f

// problem dims
#define BN   64
#define CC   256
#define KK   256
#define HH   32
#define WW   32
#define MTOT (BN*HH*WW)      // 65536
#define WELEM (KK*CC*9)      // 589824

// stats layout (floats in ws[0..127])
#define ST_GMAX 0
#define ST_M1   1
#define ST_S1   2
#define ST_M2   3
#define ST_S2   4
#define ST_PW1  8
#define ST_PA1  12
#define ST_PW2  16
#define ST_PA2  20
#define ST_IND  32          // 64 entries
#define STATS_N 128

// ws byte offsets (16B aligned)
#define WS_H1  4096
#define WS_H2  (4096 + 33554432)
#define WS_WP1 (4096 + 2*33554432)
#define WS_WP2 (WS_WP1 + 1179648)

__global__ void k_init(float* stats) {
    if (threadIdx.x < STATS_N) stats[threadIdx.x] = 0.f;
}

__global__ void k_maxabs(const float* __restrict__ w, float* stats) {
    __shared__ float red[256];
    float m = 0.f;
    for (int i = blockIdx.x * 256 + threadIdx.x; i < WELEM; i += gridDim.x * 256)
        m = fmaxf(m, fabsf(w[i]));
    red[threadIdx.x] = m;
    __syncthreads();
    for (int s = 128; s > 0; s >>= 1) {
        if (threadIdx.x < s) red[threadIdx.x] = fmaxf(red[threadIdx.x], red[threadIdx.x + s]);
        __syncthreads();
    }
    if (threadIdx.x == 0)
        atomicMax((unsigned int*)&stats[ST_GMAX], __float_as_uint(red[0]));
}

__global__ void k_indicator(const float* __restrict__ w1, const float* __restrict__ thr,
                            float* stats) {
    __shared__ float red[256];
    const int g = blockIdx.x;               // 64 groups of 4 filters
    const float* p = w1 + (size_t)g * 9216; // 4*256*9
    float s = 0.f;
    for (int i = threadIdx.x; i < 9216; i += 256) s += fabsf(p[i]);
    red[threadIdx.x] = s;
    __syncthreads();
    for (int st = 128; st > 0; st >>= 1) {
        if (threadIdx.x < st) red[threadIdx.x] += red[threadIdx.x + st];
        __syncthreads();
    }
    if (threadIdx.x == 0) {
        float gmax = stats[ST_GMAX];
        float norm = red[0] / gmax / 9216.f;
        stats[ST_IND + g] = (norm > thr[0]) ? 1.f : 0.f;
    }
}

__global__ void k_meanstd(const float* __restrict__ w1, const float* __restrict__ w2,
                          float* stats) {
    __shared__ float rs[256], rc[256];
    const int which = blockIdx.x;           // 0: w1 mask on out-ch, 1: w2 mask on in-ch
    const float* w = which ? w2 : w1;
    const int tid = threadIdx.x;
    float s = 0.f, c = 0.f;
    for (int i = tid; i < WELEM; i += 256) {
        int ch = which ? ((i / 9) & 255) : (i / 2304);
        float mk = stats[ST_IND + (ch >> 2)];
        s += w[i] * mk;
        c += mk;
    }
    rs[tid] = s; rc[tid] = c;
    __syncthreads();
    for (int st = 128; st > 0; st >>= 1) {
        if (tid < st) { rs[tid] += rs[tid + st]; rc[tid] += rc[tid + st]; }
        __syncthreads();
    }
    float mean = rs[0] / rc[0];
    float cnt  = rc[0];
    __syncthreads();
    float vv = 0.f;
    for (int i = tid; i < WELEM; i += 256) {
        int ch = which ? ((i / 9) & 255) : (i / 2304);
        float mk = stats[ST_IND + (ch >> 2)];
        float d = w[i] - mean;
        vv += mk * d * d;
    }
    rs[tid] = vv;
    __syncthreads();
    for (int st = 128; st > 0; st >>= 1) {
        if (tid < st) rs[tid] += rs[tid + st];
        __syncthreads();
    }
    if (tid == 0) {
        stats[which ? ST_M2 : ST_M1] = mean;
        stats[which ? ST_S2 : ST_S1] = sqrtf(rs[0] / (cnt - 1.f));
    }
}

__global__ void k_softmax(const float* l0, const float* l1, const float* l2, const float* l3,
                          float* stats) {
    int t = threadIdx.x;
    if (t < 4) {
        const float* L = (t == 0) ? l0 : (t == 1) ? l1 : (t == 2) ? l2 : l3;
        float a = L[0], b = L[1], c = L[2];
        float mx = fmaxf(a, fmaxf(b, c));
        float ea = __expf(a - mx), eb = __expf(b - mx), ec = __expf(c - mx);
        float s = ea + eb + ec;
        int base = 8 + 4 * t;                // PW1, PA1, PW2, PA2
        stats[base + 0] = ea / s;
        stats[base + 1] = eb / s;
        stats[base + 2] = ec / s;
    }
}

__device__ __forceinline__ float qlevel(float h, float s) {
    return rintf(h * s) / s;
}

// CDNA5 async global->LDS (ASYNCcnt-tracked), b128 per lane.
__device__ __forceinline__ void async_ld128(unsigned ldsOff, const void* gaddr) {
    asm volatile("global_load_async_to_lds_b128 %0, %1, off"
                 :: "v"(ldsOff), "v"(gaddr) : "memory");
}
__device__ __forceinline__ void wait_async0() {
    asm volatile("s_wait_asynccnt 0x0" ::: "memory");
}

// pack softmax-mixed quantized weights: wp[((tap*K + k)*C) + c] = fp16(w_mix[k][c][tap])
__global__ void k_wpack(const float* __restrict__ w1, const float* __restrict__ w2,
                        const float* __restrict__ stats,
                        _Float16* __restrict__ wp1, _Float16* __restrict__ wp2) {
    const int conv = blockIdx.y;
    const float* w = conv ? w2 : w1;
    _Float16* wp   = conv ? wp2 : wp1;
    const float mean = stats[conv ? ST_M2 : ST_M1];
    const float sd   = stats[conv ? ST_S2 : ST_S1];
    const int pb = conv ? ST_PW2 : ST_PW1;
    const float p0 = stats[pb], p1 = stats[pb + 1], p2 = stats[pb + 2];
    int o = blockIdx.x * 256 + threadIdx.x;
    if (o < WELEM) {
        int c = o & 255;
        int k = (o >> 8) & 255;
        int tap = o >> 16;
        float wv = w[k * 2304 + c * 9 + tap];
        float wn = fminf(fmaxf((wv - mean) / (sd + EPSV), -1.f), 1.f);
        float h = (wn + 1.f) * 0.5f;
        float q = p0 * (2.f * qlevel(h, 3.f)   - 1.f)
                + p1 * (2.f * qlevel(h, 15.f)  - 1.f)
                + p2 * (2.f * qlevel(h, 255.f) - 1.f);
        wp[o] = (_Float16)q;
    }
}

// x (NCHW f32) -> quant-mix(relu(bn1(x))) as fp16 NHWC, LDS-staged transpose
__global__ __launch_bounds__(256) void k_actprep(
    const float* __restrict__ x,
    const float* __restrict__ g1, const float* __restrict__ b1,
    const float* __restrict__ m1, const float* __restrict__ v1,
    const float* __restrict__ stats,
    _Float16* __restrict__ h1) {
    __shared__ _Float16 tile[32][256];
    const int by = blockIdx.x;            // b*32 + y
    const int b = by >> 5, y = by & 31;
    const float pa0 = stats[ST_PA1], pa1 = stats[ST_PA1 + 1], pa2 = stats[ST_PA1 + 2];
    for (int i = threadIdx.x; i < 8192; i += 256) {
        int c  = i >> 5;
        int xx = i & 31;
        float v = x[((size_t)(b * 256 + c)) * 1024 + y * 32 + xx];
        float inv = rsqrtf(v1[c] + EPSV);
        v = (v - m1[c]) * (g1[c] * inv) + b1[c];
        v = fmaxf(v, 0.f);
        float t = fminf(v, 1.f);
        float q = pa0 * qlevel(t, 3.f) + pa1 * qlevel(t, 15.f) + pa2 * qlevel(t, 255.f);
        tile[xx][c] = (_Float16)q;
    }
    __syncthreads();
    _Float16* dst = h1 + (size_t)by * 8192;
    for (int i = threadIdx.x; i < 8192; i += 256)
        dst[i] = tile[i >> 8][i & 255];
}

// Implicit-GEMM 3x3 SAME conv via v_wmma_f32_16x16x32_f16, double-buffered
// async global->LDS pipeline (ASYNCcnt).
// src: fp16 NHWC activations; wp: fp16 [9][K][C].
// EPI==1: fuse bn2+relu+gate+act-quant(pa2) -> fp16 NHWC (h2)
// EPI==2: add residual (NCHW f32) -> d_out (NCHW f32)
template <int EPI>
__global__ __launch_bounds__(256) void k_conv(
    const _Float16* __restrict__ src,
    const _Float16* __restrict__ wp,
    const float* __restrict__ stats,
    const float* __restrict__ g2, const float* __restrict__ b2,
    const float* __restrict__ m2, const float* __restrict__ v2,
    const float* __restrict__ xres,
    _Float16* __restrict__ dstH,
    float* __restrict__ dstF) {
    __shared__ _Float16 As[2][128 * 40];   // 128 rows x 32 halfs, pad to 40 (conflict-free b128)
    __shared__ _Float16 Bs[2][128 * 40];

    const int tid   = threadIdx.x;
    const int lane  = tid & 31;
    const int wave  = tid >> 5;
    const int waveM = wave >> 1;        // 0..3 -> 32 rows each
    const int waveN = wave & 1;         // 0..1 -> 64 cols each
    const int half  = lane >> 4;        // K-group select per ISA A/B layout
    const int l15   = lane & 15;
    const int mBase = blockIdx.x * 128;
    const int nBase = blockIdx.y * 128;
    const int NIT   = 9 * (CC / 32);    // 72 K-steps

    v8f acc[2][4];
#pragma unroll
    for (int i = 0; i < 2; ++i)
#pragma unroll
        for (int j = 0; j < 4; ++j)
#pragma unroll
            for (int e = 0; e < 8; ++e) acc[i][j][e] = 0.f;

    v8h zero8;
#pragma unroll
    for (int e = 0; e < 8; ++e) zero8[e] = (_Float16)0.f;

    // issue async global->LDS stage for K-step `it` into buffer `buf`
    auto issue = [&](int it, int buf) {
        const int tap = it >> 3;
        const int cb  = (it & 7) << 5;
        const int dy = tap / 3 - 1;
        const int dx = tap % 3 - 1;
#pragma unroll
        for (int itr = 0; itr < 2; ++itr) {
            int cid = tid + itr * 256;        // 512 chunks of 8 halfs per tile
            int row = cid >> 2;
            int cc  = (cid & 3) << 3;
            // B tile: 128 out-channels x 32 in-channels (always in-bounds)
            {
                const _Float16* pb = wp +
                    ((size_t)(tap * KK + nBase + row)) * 256 + cb + cc;
                unsigned bOff = (unsigned)(size_t)(&Bs[buf][row * 40 + cc]);
                async_ld128(bOff, pb);
            }
            // A tile: 128 positions x 32 in-channels, shifted by (dy,dx)
            {
                int m = mBase + row;
                int b = m >> 10, y = (m >> 5) & 31, xx = m & 31;
                int ys = y + dy, xs = xx + dx;
                if ((unsigned)ys < 32u && (unsigned)xs < 32u) {
                    const _Float16* pa = src +
                        ((size_t)((b << 5) + ys) * 32 + xs) * 256 + cb + cc;
                    unsigned aOff = (unsigned)(size_t)(&As[buf][row * 40 + cc]);
                    async_ld128(aOff, pa);
                } else {
                    *(v8h*)(&As[buf][row * 40 + cc]) = zero8;  // zero-pad halo
                }
            }
        }
    };

    issue(0, 0);
    for (int it = 0; it < NIT; ++it) {
        const int buf = it & 1;
        wait_async0();        // my async writes (into buf) have landed
        __syncthreads();      // everyone's landed; everyone done reading buf^1
        if (it + 1 < NIT) issue(it + 1, buf ^ 1);

        // ---- fragments per ISA 16-bit A/B layout ----
        v16h afr[2], bfr[4];
#pragma unroll
        for (int mf = 0; mf < 2; ++mf) {
            const _Float16* p = &As[buf][(waveM * 32 + mf * 16 + l15) * 40 + half * 8];
            union { v16h v; v8h h[2]; } u;
            u.h[0] = *(const v8h*)p;         // K = klo..klo+7
            u.h[1] = *(const v8h*)(p + 16);  // K = klo+16..klo+23
            afr[mf] = u.v;
        }
#pragma unroll
        for (int nf = 0; nf < 4; ++nf) {
            const _Float16* p = &Bs[buf][(waveN * 64 + nf * 16 + l15) * 40 + half * 8];
            union { v16h v; v8h h[2]; } u;
            u.h[0] = *(const v8h*)p;
            u.h[1] = *(const v8h*)(p + 16);
            bfr[nf] = u.v;
        }
#pragma unroll
        for (int mf = 0; mf < 2; ++mf)
#pragma unroll
            for (int nf = 0; nf < 4; ++nf)
                acc[mf][nf] = __builtin_amdgcn_wmma_f32_16x16x32_f16(
                    false, afr[mf], false, bfr[nf],
                    (short)0, acc[mf][nf], false, false);
    }

    // ---- epilogue ----
    float pa0 = 0.f, pa1 = 0.f, pa2 = 0.f;
    if (EPI == 1) {
        pa0 = stats[ST_PA2]; pa1 = stats[ST_PA2 + 1]; pa2 = stats[ST_PA2 + 2];
    }
#pragma unroll
    for (int nf = 0; nf < 4; ++nf) {
        int N = nBase + waveN * 64 + nf * 16 + l15;
        float sc = 0.f, bi = 0.f, mu = 0.f, gate = 0.f;
        if (EPI == 1) {
            float inv = rsqrtf(v2[N] + EPSV);
            sc = g2[N] * inv; bi = b2[N]; mu = m2[N];
            gate = stats[ST_IND + (N >> 2)];
        }
#pragma unroll
        for (int mf = 0; mf < 2; ++mf) {
#pragma unroll
            for (int r = 0; r < 8; ++r) {
                int M = mBase + waveM * 32 + mf * 16 + half * 8 + r;
                float v = acc[mf][nf][r];
                if (EPI == 1) {
                    float t = (v - mu) * sc + bi;
                    t = fmaxf(t, 0.f) * gate;
                    float tc = fminf(fmaxf(t, 0.f), 1.f);
                    float q = pa0 * qlevel(tc, 3.f) + pa1 * qlevel(tc, 15.f)
                            + pa2 * qlevel(tc, 255.f);
                    dstH[(size_t)M * 256 + N] = (_Float16)q;
                } else {
                    int b = M >> 10, y = (M >> 5) & 31, xx = M & 31;
                    size_t o = ((size_t)(b * 256 + N) * 32 + y) * 32 + xx;
                    dstF[o] = v + xres[o];
                }
            }
        }
    }
}

extern "C" void kernel_launch(void* const* d_in, const int* in_sizes, int n_in,
                              void* d_out, int out_size, void* d_ws, size_t ws_size,
                              hipStream_t stream) {
    const float* x   = (const float*)d_in[0];
    const float* w1  = (const float*)d_in[1];
    const float* w2  = (const float*)d_in[2];
    const float* g1  = (const float*)d_in[3];
    const float* b1  = (const float*)d_in[4];
    const float* m1  = (const float*)d_in[5];
    const float* v1  = (const float*)d_in[6];
    const float* g2  = (const float*)d_in[7];
    const float* b2  = (const float*)d_in[8];
    const float* m2  = (const float*)d_in[9];
    const float* v2  = (const float*)d_in[10];
    const float* thr = (const float*)d_in[11];
    const float* wl1 = (const float*)d_in[12];
    const float* al1 = (const float*)d_in[13];
    const float* wl2 = (const float*)d_in[14];
    const float* al2 = (const float*)d_in[15];
    float* out = (float*)d_out;

    char* ws = (char*)d_ws;
    float*    stats = (float*)ws;
    _Float16* h1    = (_Float16*)(ws + WS_H1);
    _Float16* h2    = (_Float16*)(ws + WS_H2);
    _Float16* wp1   = (_Float16*)(ws + WS_WP1);
    _Float16* wp2   = (_Float16*)(ws + WS_WP2);

    k_init<<<1, 128, 0, stream>>>(stats);
    k_maxabs<<<256, 256, 0, stream>>>(w1, stats);
    k_indicator<<<64, 256, 0, stream>>>(w1, thr, stats);
    k_meanstd<<<2, 256, 0, stream>>>(w1, w2, stats);
    k_softmax<<<1, 32, 0, stream>>>(wl1, al1, wl2, al2, stats);
    k_wpack<<<dim3((WELEM + 255) / 256, 2), 256, 0, stream>>>(w1, w2, stats, wp1, wp2);
    k_actprep<<<BN * HH, 256, 0, stream>>>(x, g1, b1, m1, v1, stats, h1);

    dim3 cgrid(MTOT / 128, KK / 128);   // 512 x 2
    k_conv<1><<<cgrid, 256, 0, stream>>>(h1, wp1, stats, g2, b2, m2, v2,
                                         nullptr, h2, nullptr);
    k_conv<2><<<cgrid, 256, 0, stream>>>(h2, wp2, stats, nullptr, nullptr, nullptr, nullptr,
                                         x, nullptr, out);
}
